// TransEncoder_87986700026318
// MI455X (gfx1250) — matrix operand order repaired and checked by reading, not compiled, not run on hardware
//
#include <hip/hip_runtime.h>

// ---------------- problem constants (from reference) ----------------
#define NNODES 10000
#define NEDGES 320000
#define HID    128
#define HEADS  4
#define DQKV   512        // HEADS*HID
#define IN_DIM 256

typedef _Float16 half8  __attribute__((ext_vector_type(8)));
typedef _Float16 half16 __attribute__((ext_vector_type(16)));
typedef float    float8 __attribute__((ext_vector_type(8)));

// ---------------- small utility kernels ----------------
__global__ void to_f16_kernel(const float* __restrict__ in, _Float16* __restrict__ out, int n) {
  int i = blockIdx.x * blockDim.x + threadIdx.x;
  if (i < n) out[i] = (_Float16)in[i];
}

__global__ void fill_u32_kernel(unsigned* __restrict__ p, unsigned v, int n) {
  int i = blockIdx.x * blockDim.x + threadIdx.x;
  if (i < n) p[i] = v;
}

// monotone float<->unsigned key (unsigned order == float order)
__device__ __forceinline__ unsigned fkey(float f) {
  unsigned b = __float_as_uint(f);
  return (b & 0x80000000u) ? ~b : (b | 0x80000000u);
}
__device__ __forceinline__ float fkey_inv(unsigned k) {
  unsigned b = (k & 0x80000000u) ? (k & 0x7fffffffu) : ~k;
  return __uint_as_float(b);
}

// ---------------- WMMA GEMM: C[M,Nd] = A[M,K] (f16) * B[K,Nd] (f16) + bias ----------------
// One 16x64 block per wave (4 accumulators share one A fragment -> 4 WMMA per A load).
// 8 waves per block. M%16==0, K%32==0, Nd%64==0 guaranteed here.
// Layouts per CDNA5 ISA 7.12.2:
//   A (16x32 f16): lane<16 -> row M=lane, halves 0..7 = K 0..7, halves 8..15 = K 16..23
//                  lane>=16 -> same row group, K 8..15 and 24..31
//   B (32x16 f16): lane&15 -> col N, halves 0..7 = K 8*(lane>>4)+0..7, halves 8..15 = K 16+8*(lane>>4)+0..7
//   C/D (16x16 f32): lane<16 -> N=lane, vgpr r -> M=r ; lane>=16 -> N=lane-16, M=8+r
__global__ void __launch_bounds__(256)
gemm_f16_wmma_kernel(const _Float16* __restrict__ A, const _Float16* __restrict__ B,
                     const float* __restrict__ bias, float* __restrict__ C,
                     int K, int Nd, int NG /* Nd/64 */, int totalTasks)
{
  int t = blockIdx.x * 8 + (threadIdx.x >> 5);
  if (t >= totalTasks) return;            // wave-uniform exit: EXEC all-ones for WMMA
  int lane = threadIdx.x & 31;
  int m0 = (t / NG) << 4;
  int n0 = (t % NG) << 6;                 // 64-wide column group
  int lm = lane & 15;
  int hi = lane >> 4;                     // 0 or 1

  const _Float16* Arow  = A + (size_t)(m0 + lm) * K + (hi << 3);
  const _Float16* Bbase = B + n0 + lm;

  float8 z = {0.f, 0.f, 0.f, 0.f, 0.f, 0.f, 0.f, 0.f};
  float8 acc0 = z, acc1 = z, acc2 = z, acc3 = z;

  for (int k0 = 0; k0 < K; k0 += 32) {
    half8 a_lo = *(const half8*)(Arow + k0);        // K: k0 + hi*8 + [0..7]
    half8 a_hi = *(const half8*)(Arow + k0 + 16);   // K: k0+16 + hi*8 + [0..7]
    half16 a = __builtin_shufflevector(a_lo, a_hi,
        0, 1, 2, 3, 4, 5, 6, 7, 8, 9, 10, 11, 12, 13, 14, 15);
    const _Float16* Bk = Bbase + (size_t)k0 * Nd;
    half16 b0, b1, b2, b3;
#pragma unroll
    for (int s = 0; s < 16; ++s) {
      size_t kr = (size_t)((s < 8 ? s : s + 8) + (hi << 3)) * Nd;
      b0[s] = Bk[kr];
      b1[s] = Bk[kr + 16];
      b2[s] = Bk[kr + 32];
      b3[s] = Bk[kr + 48];
    }
    acc0 = __builtin_amdgcn_wmma_f32_16x16x32_f16(false, a, false, b0, (short)0, acc0, false, false);
    acc1 = __builtin_amdgcn_wmma_f32_16x16x32_f16(false, a, false, b1, (short)0, acc1, false, false);
    acc2 = __builtin_amdgcn_wmma_f32_16x16x32_f16(false, a, false, b2, (short)0, acc2, false, false);
    acc3 = __builtin_amdgcn_wmma_f32_16x16x32_f16(false, a, false, b3, (short)0, acc3, false, false);
  }

  int row = m0 + hi * 8;
  float8 accs[4] = {acc0, acc1, acc2, acc3};
#pragma unroll
  for (int j = 0; j < 4; ++j) {
    int col = n0 + j * 16 + lm;
    float bv = bias[col];
    float* Cp = C + (size_t)row * Nd + col;
#pragma unroll
    for (int r = 0; r < 8; ++r) Cp[(size_t)r * Nd] = accs[j][r] + bv;
  }
}

// ---------------- per-edge attention logits: alpha[e,h] = <q[dst,h,:],k[src,h,:]>/sqrt(HID) ----
// wave per edge; lane covers 16 contiguous channels; heads = groups of 8 lanes
__global__ void __launch_bounds__(256)
edge_logits_kernel(const float* __restrict__ q, const float* __restrict__ k,
                   const int* __restrict__ src, const int* __restrict__ dst,
                   float* __restrict__ alpha, int E)
{
  int e = blockIdx.x * 8 + (threadIdx.x >> 5);
  if (e >= E) return;
  int lane = threadIdx.x & 31;
  int s = src[e], d = dst[e];
  const float4* qp = (const float4*)(q + (size_t)d * DQKV) + lane * 4;
  const float4* kp = (const float4*)(k + (size_t)s * DQKV) + lane * 4;
  float part = 0.f;
#pragma unroll
  for (int j = 0; j < 4; ++j) {
    float4 a = qp[j], b = kp[j];
    part += a.x * b.x + a.y * b.y + a.z * b.z + a.w * b.w;
  }
  part += __shfl_xor(part, 1, 32);
  part += __shfl_xor(part, 2, 32);
  part += __shfl_xor(part, 4, 32);
  if ((lane & 7) == 0)
    alpha[(size_t)e * HEADS + (lane >> 3)] = part * 0.08838834764831845f; // 1/sqrt(128)
}

// ---------------- segment max over incoming edges (per dst,head) ----------------
__global__ void edge_max_kernel(const float* __restrict__ alpha, const int* __restrict__ dst,
                                unsigned* __restrict__ umax, int EH)
{
  int i = blockIdx.x * blockDim.x + threadIdx.x;
  if (i >= EH) return;
  int e = i >> 2, h = i & 3;
  atomicMax(&umax[dst[e] * HEADS + h], fkey(alpha[i]));
}

// ---------------- a = exp(alpha - max); segment sum ----------------
__global__ void edge_expsum_kernel(float* __restrict__ alpha, const int* __restrict__ dst,
                                   const unsigned* __restrict__ umax, float* __restrict__ sums,
                                   int EH)
{
  int i = blockIdx.x * blockDim.x + threadIdx.x;
  if (i >= EH) return;
  int e = i >> 2, h = i & 3;
  int d = dst[e];
  float a = __expf(alpha[i] - fkey_inv(umax[d * HEADS + h]));
  alpha[i] = a;
  atomicAdd(&sums[d * HEADS + h], a);
}

// ---------------- accum[dst,c] += sum_h w_h * v[src,h*HID+c]  (wave per edge) -----------
// Head-mean is linear, so combine the 4 heads in-register: only 128 atomics per edge.
__global__ void __launch_bounds__(256)
edge_aggregate_kernel(const float* __restrict__ alpha, const float* __restrict__ sums,
                      const float* __restrict__ v, const int* __restrict__ src,
                      const int* __restrict__ dst, float* __restrict__ accum, int E)
{
  int e = blockIdx.x * 8 + (threadIdx.x >> 5);
  if (e >= E) return;
  int lane = threadIdx.x & 31;
  int s = src[e], d = dst[e];
  float4 a4 = *(const float4*)(alpha + (size_t)e * HEADS);
  float4 s4 = *(const float4*)(sums + (size_t)d * HEADS);
  float w0 = a4.x / (s4.x + 1e-16f);
  float w1 = a4.y / (s4.y + 1e-16f);
  float w2 = a4.z / (s4.z + 1e-16f);
  float w3 = a4.w / (s4.w + 1e-16f);
  const float* vrow = v + (size_t)s * DQKV + lane * 4;   // 4 channels per lane
  float4 v0 = *(const float4*)(vrow);
  float4 v1 = *(const float4*)(vrow + HID);
  float4 v2 = *(const float4*)(vrow + 2 * HID);
  float4 v3 = *(const float4*)(vrow + 3 * HID);
  float c0 = w0 * v0.x + w1 * v1.x + w2 * v2.x + w3 * v3.x;
  float c1 = w0 * v0.y + w1 * v1.y + w2 * v2.y + w3 * v3.y;
  float c2 = w0 * v0.z + w1 * v1.z + w2 * v2.z + w3 * v3.z;
  float c3 = w0 * v0.w + w1 * v1.w + w2 * v2.w + w3 * v3.w;
  float* op = accum + (size_t)d * HID + lane * 4;
  atomicAdd(op + 0, c0);
  atomicAdd(op + 1, c1);
  atomicAdd(op + 2, c2);
  atomicAdd(op + 3, c3);
}

// ---------------- out = relu(0.25*accum + skip) ----------------
__global__ void finalize_kernel(const float* __restrict__ accum, const float* __restrict__ skip,
                                float* __restrict__ out, int NH)
{
  int i = blockIdx.x * blockDim.x + threadIdx.x;
  if (i >= NH) return;
  out[i] = fmaxf(0.25f * accum[i] + skip[i], 0.f);
}

// ---------------- host-side orchestration ----------------
struct Ws {
  _Float16 *h16, *wq16, *wk16, *wv16, *ws16;
  float *q, *k, *v, *s, *alpha, *sums, *accum, *hnext;
  unsigned* umax;
};

static inline int cdiv(int a, int b) { return (a + b - 1) / b; }

static void run_layer(const float* h, int din,
                      const float* Wq, const float* bq, const float* Wk, const float* bk,
                      const float* Wv, const float* bv, const float* Wskip, const float* bs,
                      const int* src, const int* dst, float* out, const Ws& w,
                      hipStream_t stream)
{
  const int N = NNODES, E = NEDGES;
  // 1) convert activations + weights to f16
  to_f16_kernel<<<cdiv(N * din, 256), 256, 0, stream>>>(h, w.h16, N * din);
  to_f16_kernel<<<cdiv(din * DQKV, 256), 256, 0, stream>>>(Wq, w.wq16, din * DQKV);
  to_f16_kernel<<<cdiv(din * DQKV, 256), 256, 0, stream>>>(Wk, w.wk16, din * DQKV);
  to_f16_kernel<<<cdiv(din * DQKV, 256), 256, 0, stream>>>(Wv, w.wv16, din * DQKV);
  to_f16_kernel<<<cdiv(din * HID, 256), 256, 0, stream>>>(Wskip, w.ws16, din * HID);

  // 2) WMMA GEMMs: one 16x64 block per wave (N=10000 -> 625 row tiles exactly)
  int mt = N / 16;
  int ngQ = DQKV / 64, ngS = HID / 64;
  int tasksQ = mt * ngQ, tasksS = mt * ngS;
  gemm_f16_wmma_kernel<<<cdiv(tasksQ, 8), 256, 0, stream>>>(w.h16, w.wq16, bq, w.q, din, DQKV, ngQ, tasksQ);
  gemm_f16_wmma_kernel<<<cdiv(tasksQ, 8), 256, 0, stream>>>(w.h16, w.wk16, bk, w.k, din, DQKV, ngQ, tasksQ);
  gemm_f16_wmma_kernel<<<cdiv(tasksQ, 8), 256, 0, stream>>>(w.h16, w.wv16, bv, w.v, din, DQKV, ngQ, tasksQ);
  gemm_f16_wmma_kernel<<<cdiv(tasksS, 8), 256, 0, stream>>>(w.h16, w.ws16, bs, w.s, din, HID, ngS, tasksS);

  // 3) edge logits
  edge_logits_kernel<<<cdiv(E, 8), 256, 0, stream>>>(w.q, w.k, src, dst, w.alpha, E);

  // 4) segment softmax over dst
  fill_u32_kernel<<<cdiv(N * HEADS, 256), 256, 0, stream>>>(w.umax, 0u, N * HEADS);
  edge_max_kernel<<<cdiv(E * HEADS, 256), 256, 0, stream>>>(w.alpha, dst, w.umax, E * HEADS);
  fill_u32_kernel<<<cdiv(N * HEADS, 256), 256, 0, stream>>>((unsigned*)w.sums, 0u, N * HEADS);
  edge_expsum_kernel<<<cdiv(E * HEADS, 256), 256, 0, stream>>>(w.alpha, dst, w.umax, w.sums, E * HEADS);

  // 5) scatter-aggregate with heads pre-combined (accum aliases q; q dead after logits)
  fill_u32_kernel<<<cdiv(N * HID, 256), 256, 0, stream>>>((unsigned*)w.accum, 0u, N * HID);
  edge_aggregate_kernel<<<cdiv(E, 8), 256, 0, stream>>>(w.alpha, w.sums, w.v, src, dst, w.accum, E);

  // 6) head-mean + skip + relu
  finalize_kernel<<<cdiv(N * HID, 256), 256, 0, stream>>>(w.accum, w.s, out, N * HID);
}

extern "C" void kernel_launch(void* const* d_in, const int* in_sizes, int n_in,
                              void* d_out, int out_size, void* d_ws, size_t ws_size,
                              hipStream_t stream)
{
  const int N = NNODES, E = NEDGES;
  const float* x  = (const float*)d_in[0];
  const int* ei   = (const int*)d_in[1];   // (2, E) int32: row0=src, row1=dst
  const int* src  = ei;
  const int* dst  = ei + E;
  // edge_attr (d_in[2]) is unused by the reference (edge_dim=None)

  const float *Wq0 = (const float*)d_in[3],  *bq0 = (const float*)d_in[4];
  const float *Wk0 = (const float*)d_in[5],  *bk0 = (const float*)d_in[6];
  const float *Wv0 = (const float*)d_in[7],  *bv0 = (const float*)d_in[8];
  const float *Ws0 = (const float*)d_in[9],  *bs0 = (const float*)d_in[10];
  const float *Wq1 = (const float*)d_in[11], *bq1 = (const float*)d_in[12];
  const float *Wk1 = (const float*)d_in[13], *bk1 = (const float*)d_in[14];
  const float *Wv1 = (const float*)d_in[15], *bv1 = (const float*)d_in[16];
  const float *Ws1 = (const float*)d_in[17], *bs1 = (const float*)d_in[18];

  // carve scratch (all offsets 256B aligned; total ~78 MB)
  size_t off = 0;
  auto take = [&](size_t bytes) -> void* {
    void* p = (void*)((char*)d_ws + off);
    off += (bytes + 255) & ~(size_t)255;
    return p;
  };
  Ws w;
  w.h16   = (_Float16*)take((size_t)N * IN_DIM * 2);
  w.wq16  = (_Float16*)take((size_t)IN_DIM * DQKV * 2);
  w.wk16  = (_Float16*)take((size_t)IN_DIM * DQKV * 2);
  w.wv16  = (_Float16*)take((size_t)IN_DIM * DQKV * 2);
  w.ws16  = (_Float16*)take((size_t)IN_DIM * HID * 2);
  w.q     = (float*)take((size_t)N * DQKV * 4);
  w.k     = (float*)take((size_t)N * DQKV * 4);
  w.v     = (float*)take((size_t)N * DQKV * 4);
  w.s     = (float*)take((size_t)N * HID * 4);
  w.alpha = (float*)take((size_t)E * HEADS * 4);
  w.umax  = (unsigned*)take((size_t)N * HEADS * 4);
  w.sums  = (float*)take((size_t)N * HEADS * 4);
  w.accum = w.q;                 // alias: q dead after edge_logits (only N*HID used)
  w.hnext = w.k;                 // alias: k dead after edge_logits

  // layer 0: x (N x 256) -> hnext (N x 128)
  run_layer(x, IN_DIM, Wq0, bq0, Wk0, bk0, Wv0, bv0, Ws0, bs0, src, dst, w.hnext, w, stream);
  // layer 1: hnext (N x 128) -> d_out (N x 128)
  run_layer(w.hnext, HID, Wq1, bq1, Wk1, bk1, Wv1, bv1, Ws1, bs1, src, dst,
            (float*)d_out, w, stream);
}
